// CrossAttention_68341519614427
// MI455X (gfx1250) — matrix operand order repaired and checked by reading, not compiled, not run on hardware
//
#include <hip/hip_runtime.h>
#include <hip/hip_bf16.h>

typedef __bf16 bf16_t;
typedef __attribute__((ext_vector_type(16))) __bf16 v16bf;
typedef __attribute__((ext_vector_type(8)))  __bf16 v8bf;
typedef __attribute__((ext_vector_type(4)))  __bf16 v4bf;
typedef __attribute__((ext_vector_type(8)))  float  v8f;
typedef __attribute__((ext_vector_type(4)))  float  v4f;

// ---------------------------------------------------------------------------
// WMMA helpers (CDNA5 16x16x32 bf16, f32 accumulate). Layouts per ISA 7.12.2:
//  A (16x32, 16-bit): lane L holds row L%16; elem i (0..7) = K(kh*8+i),
//                     elem 8+i = K(16+kh*8+i), kh = L/16.
//  B (32x16, 16-bit): lane L holds col L%16; elem i (0..7) = K(kh*8+i), elem 8+i = K(16+kh*8+i).
//  C/D (16x16 f32):   elem v = row(v + 8*(L/16)), col L%16.
// ---------------------------------------------------------------------------
__device__ __forceinline__ v8f wmma_bf16(v16bf a, v16bf b, v8f c) {
  return __builtin_amdgcn_wmma_f32_16x16x32_bf16(false, a, false, b,
                                                 (short)0, c, false, false);
}

// A-fragment from LDS tile stored row-major [16][>=32], row stride ld (elems).
__device__ __forceinline__ v16bf lds_load_a(const bf16_t* tile, int ld) {
  const int lane = threadIdx.x & 31;
  const int row  = lane & 15;
  const int kh   = lane >> 4;
  const bf16_t* p = tile + row * ld + kh * 8;
  v8bf lo = *(const v8bf*)p;            // K = kh*8 + 0..7
  v8bf hi = *(const v8bf*)(p + 16);     // K = 16 + kh*8 + 0..7
  return __builtin_shufflevector(lo, hi, 0,1,2,3,4,5,6,7,
                                         8,9,10,11,12,13,14,15);
}

// B-fragment: tile stored as [n][k] row-major (so frag reads are contiguous),
// row stride ld (elems). Fragment covers K=0..31 of the tile's k axis.
__device__ __forceinline__ v16bf lds_load_b(const bf16_t* tile, int ld) {
  const int lane = threadIdx.x & 31;
  const int col  = lane & 15;
  const int kh   = lane >> 4;
  const bf16_t* p = tile + col * ld + kh * 8;
  v8bf lo = *(const v8bf*)p;            // K = kh*8 + 0..7
  v8bf hi = *(const v8bf*)(p + 16);     // K = 16 + kh*8 + 0..7
  return __builtin_shufflevector(lo, hi, 0,1,2,3,4,5,6,7,
                                         8,9,10,11,12,13,14,15);
}

// ---------------------------------------------------------------------------
// CDNA5 async global->LDS copy (16B per lane), tracked by ASYNCcnt.
// ISA 10.2: low 32 bits of a generic LDS pointer == LDS byte address, so we
// pass the truncated pointer as vdst. GV mode: 64-bit per-lane vaddr, no saddr.
// ---------------------------------------------------------------------------
__device__ __forceinline__ void async_copy16(void* lds_dst, const void* gsrc) {
  unsigned lds_off = (unsigned)(uintptr_t)lds_dst;
  unsigned long long ga = (unsigned long long)(uintptr_t)gsrc;
  asm volatile("global_load_async_to_lds_b128 %0, %1, off"
               :: "v"(lds_off), "v"(ga) : "memory");
}
__device__ __forceinline__ void wait_async0() {
  asm volatile("s_wait_asynccnt 0" ::: "memory");
}

// ---------------------------------------------------------------------------
// Projection GEMM: Y[m][n] = sum_k X[m][k] * W[n][k] + bias[n], Y in bf16.
// Block tile: 128(M) x 64(N), K chunks of 32. 256 threads = 8 waves,
// wave w owns rows [w*16, w*16+16), 4 col-tiles => 4 v8f accumulators.
// (f32 -> bf16 conversion forces the register staging path here.)
// ---------------------------------------------------------------------------
__global__ __launch_bounds__(256)
void proj_kernel(const float* __restrict__ X, const float* __restrict__ W,
                 const float* __restrict__ bias, bf16_t* __restrict__ Y,
                 int Dout, int K) {
  constexpr int LDA = 48;   // padded row stride (elems) for X tile
  constexpr int LDB = 40;   // padded row stride (elems) for W tile
  __shared__ __align__(16) bf16_t As[128 * LDA];
  __shared__ __align__(16) bf16_t Ws[64 * LDB];

  const int tid  = threadIdx.x;
  const int wave = tid >> 5;
  const int lane = tid & 31;
  const int m0   = blockIdx.x * 128;
  const int n0   = blockIdx.y * 64;

  v8f acc[4] = {};

  for (int kc = 0; kc < K; kc += 32) {
    // Stage X tile 128x32 (f32 -> bf16): 1024 float4 / 256 threads.
#pragma unroll
    for (int r = 0; r < 4; ++r) {
      int idx = r * 256 + tid;
      int row = idx >> 3, c4 = idx & 7;
      v4f x = *(const v4f*)(X + (size_t)(m0 + row) * K + kc + c4 * 4);
      v4bf xb = { (bf16_t)x[0], (bf16_t)x[1], (bf16_t)x[2], (bf16_t)x[3] };
      *(v4bf*)(As + row * LDA + c4 * 4) = xb;
    }
    // Stage W tile 64x32 (rows n, cols k): 512 float4 / 256 threads.
#pragma unroll
    for (int r = 0; r < 2; ++r) {
      int idx = r * 256 + tid;
      int row = idx >> 3, c4 = idx & 7;
      v4f w = *(const v4f*)(W + (size_t)(n0 + row) * K + kc + c4 * 4);
      v4bf wb = { (bf16_t)w[0], (bf16_t)w[1], (bf16_t)w[2], (bf16_t)w[3] };
      *(v4bf*)(Ws + row * LDB + c4 * 4) = wb;
    }
    __syncthreads();

    v16bf a = lds_load_a(As + wave * 16 * LDA, LDA);
#pragma unroll
    for (int nt = 0; nt < 4; ++nt) {
      v16bf b = lds_load_b(Ws + nt * 16 * LDB, LDB);
      acc[nt] = wmma_bf16(a, b, acc[nt]);
    }
    __syncthreads();
  }

  // Epilogue: bias add, bf16 store.
  const int col  = lane & 15;
  const int half = lane >> 4;
#pragma unroll
  for (int nt = 0; nt < 4; ++nt) {
    float bv = bias[n0 + nt * 16 + col];
#pragma unroll
    for (int v = 0; v < 8; ++v) {
      int row = m0 + wave * 16 + half * 8 + v;
      Y[(size_t)row * Dout + n0 + nt * 16 + col] = (bf16_t)(acc[nt][v] + bv);
    }
  }
}

// ---------------------------------------------------------------------------
// Flash attention: one block = (b, h, 64 query rows). 128 threads = 4 waves,
// wave owns 16 query rows. Streams 64-key KV tiles through double-buffered
// LDS; K tiles move via async global->LDS (ASYNCcnt), V is transposed in
// registers. Next tile's loads are issued before current tile's compute.
// Output written in reference's flat order: out[b*N*D + h*N*pd + n*pd + p].
// ---------------------------------------------------------------------------
constexpr int Nn = 2048, D = 768, H = 12, PD = 64, QB = 64, KB = 64;
constexpr int LD = 72;  // padded row stride (elems)

__device__ __forceinline__ void load_kv_tile(const bf16_t* __restrict__ KVb,
                                             int b, int h, int j0,
                                             bf16_t* Ks, bf16_t* Vt, int tid) {
#pragma unroll
  for (int r = 0; r < 4; ++r) {
    int idx = r * 128 + tid;
    int row = idx >> 3, seg = idx & 7;
    size_t base = (size_t)(b * Nn + j0 + row) * (2 * D) + h * PD + seg * 8;
    // K rows: direct async 16B global->LDS copy (no conversion needed).
    async_copy16(Ks + row * LD + seg * 8, KVb + base);
    // V rows: load + transpose into Vt[p][j] so PV B-frags read contiguous.
    v8bf v8_ = *(const v8bf*)(KVb + base + D);
#pragma unroll
    for (int e = 0; e < 8; ++e)
      Vt[(seg * 8 + e) * LD + row] = v8_[e];
  }
}

__global__ __launch_bounds__(128)
void attn_kernel(const bf16_t* __restrict__ Qb, const bf16_t* __restrict__ KVb,
                 float* __restrict__ Out) {
  __shared__ __align__(16) bf16_t Qs[QB * LD];         // [qrow][p]
  __shared__ __align__(16) bf16_t Ks[2][KB * LD];      // [jrow][p] double buf
  __shared__ __align__(16) bf16_t Vt[2][PD * LD];      // [p][jrow] double buf
  __shared__ __align__(16) bf16_t Ps[4][16 * LD];      // per-wave P [row][j]

  const int blk = blockIdx.x;
  const int qb  = blk & 31;            // N/QB = 32
  const int h   = (blk >> 5) % H;
  const int b   = blk / (32 * H);
  const int q0  = qb * QB;

  const int tid  = threadIdx.x;
  const int wave = tid >> 5;
  const int lane = tid & 31;
  const int col  = lane & 15;
  const int half = lane >> 4;

  // Stage Q tile 64x64 bf16 via async copies, plus first KV tile.
#pragma unroll
  for (int r = 0; r < 4; ++r) {
    int idx = r * 128 + tid;           // 512 v8bf
    int row = idx >> 3, seg = idx & 7;
    async_copy16(Qs + row * LD + seg * 8,
                 Qb + (size_t)(b * Nn + q0 + row) * D + h * PD + seg * 8);
  }
  load_kv_tile(KVb, b, h, 0, Ks[0], Vt[0], tid);
  wait_async0();
  __syncthreads();

  v16bf qf0 = lds_load_a(Qs + wave * 16 * LD, LD);
  v16bf qf1 = lds_load_a(Qs + wave * 16 * LD + 32, LD);

  v8f acc[4] = {};
  float mcur[8], lsum[8];
#pragma unroll
  for (int v = 0; v < 8; ++v) { mcur[v] = -__builtin_inff(); lsum[v] = 0.f; }

  const float L2E = 1.4426950408889634f;

  for (int j0 = 0; j0 < Nn; j0 += KB) {
    const int cur = (j0 / KB) & 1;
    // Kick off next tile's loads first; they overlap the compute below.
    if (j0 + KB < Nn)
      load_kv_tile(KVb, b, h, j0 + KB, Ks[cur ^ 1], Vt[cur ^ 1], tid);

    // S = Q K^T  (4 j-tiles x 2 K-chunks = 8 WMMAs)
    v8f s[4];
#pragma unroll
    for (int jt = 0; jt < 4; ++jt) {
      v8f z = {};
      z = wmma_bf16(qf0, lds_load_b(Ks[cur] + jt * 16 * LD, LD), z);
      z = wmma_bf16(qf1, lds_load_b(Ks[cur] + jt * 16 * LD + 32, LD), z);
      s[jt] = z;
    }

    // Online softmax (row stats via 16-lane xor reductions).
#pragma unroll
    for (int v = 0; v < 8; ++v) {
      float mx = fmaxf(fmaxf(s[0][v], s[1][v]), fmaxf(s[2][v], s[3][v]));
#pragma unroll
      for (int off = 8; off >= 1; off >>= 1)
        mx = fmaxf(mx, __shfl_xor(mx, off, 32));
      float mnew  = fmaxf(mcur[v], mx);
      float alpha = __builtin_exp2f((mcur[v] - mnew) * L2E);
      mcur[v] = mnew;
      float ps = 0.f;
#pragma unroll
      for (int jt = 0; jt < 4; ++jt) {
        float p = __builtin_exp2f((s[jt][v] - mnew) * L2E);
        ps += p;
        Ps[wave][(half * 8 + v) * LD + jt * 16 + col] = (bf16_t)p;
      }
#pragma unroll
      for (int off = 8; off >= 1; off >>= 1)
        ps += __shfl_xor(ps, off, 32);
      lsum[v] = lsum[v] * alpha + ps;
#pragma unroll
      for (int pt = 0; pt < 4; ++pt) acc[pt][v] *= alpha;
    }

    // O += P V  (2 j-chunks x 4 p-tiles = 8 WMMAs). Wave-local P round-trip;
    // in-wave DS dependencies are resolved by compiler-inserted s_wait_dscnt.
#pragma unroll
    for (int kc = 0; kc < 2; ++kc) {
      v16bf pa = lds_load_a(&Ps[wave][kc * 32], LD);
#pragma unroll
      for (int pt = 0; pt < 4; ++pt) {
        v16bf vb = lds_load_b(Vt[cur] + pt * 16 * LD + kc * 32, LD);
        acc[pt] = wmma_bf16(pa, vb, acc[pt]);
      }
    }

    // Async K copies + ds V stores for the next buffer must be LDS-visible
    // before any wave reads it after this barrier.
    wait_async0();
    __syncthreads();
  }

  // Final normalize + store, reference flat order (no head permute).
  size_t outbase = (size_t)b * Nn * D + (size_t)h * Nn * PD + (size_t)q0 * PD;
#pragma unroll
  for (int v = 0; v < 8; ++v) {
    int   row = wave * 16 + half * 8 + v;
    float inv = 1.0f / lsum[v];
#pragma unroll
    for (int pt = 0; pt < 4; ++pt)
      Out[outbase + (size_t)row * PD + pt * 16 + col] = acc[pt][v] * inv;
  }
}

// ---------------------------------------------------------------------------
extern "C" void kernel_launch(void* const* d_in, const int* in_sizes, int n_in,
                              void* d_out, int out_size, void* d_ws, size_t ws_size,
                              hipStream_t stream) {
  const float* x1  = (const float*)d_in[0];
  const float* x2  = (const float*)d_in[1];
  const float* Wq  = (const float*)d_in[2];
  const float* bq  = (const float*)d_in[3];
  const float* Wkv = (const float*)d_in[4];
  const float* bkv = (const float*)d_in[5];
  float* out = (float*)d_out;

  constexpr int B = 2, N = 2048, Dd = 768;
  constexpr int M = B * N;  // 4096

  bf16_t* qbf  = (bf16_t*)d_ws;                   // (M, D)    bf16
  bf16_t* kvbf = qbf + (size_t)M * Dd;            // (M, 2D)   bf16

  dim3 gq(M / 128, Dd / 64);        // 32 x 12
  proj_kernel<<<gq, 256, 0, stream>>>(x2, Wq, bq, qbf, Dd, Dd);

  dim3 gkv(M / 128, (2 * Dd) / 64); // 32 x 24
  proj_kernel<<<gkv, 256, 0, stream>>>(x1, Wkv, bkv, kvbf, 2 * Dd, Dd);

  attn_kernel<<<B * 12 * (N / 64), 128, 0, stream>>>(qbf, kvbf, out);
}
